// YOLOXHead_86517821215098
// MI455X (gfx1250) — compile-verified, matrix operand order checked by reading, not run on hardware
//
#include <hip/hip_runtime.h>
#include <hip/hip_bf16.h>

typedef __bf16 bf16;
typedef __attribute__((ext_vector_type(16))) __bf16 v16bf;
typedef __attribute__((ext_vector_type(8)))  __bf16 v8bf;
typedef __attribute__((ext_vector_type(8)))  float  v8f;

// ---------------- workspace layout (bytes) ----------------
static constexpr size_t ACT_BYTES = (size_t)8 * 80 * 80 * 256 * 2;   // 26,214,400 (largest level)
static constexpr size_t OFF_IN    = 0;                                // bf16 NHWC input (reused per level)
static constexpr size_t OFF_X     = OFF_IN + ACT_BYTES;               // stem output
static constexpr size_t OFF_T     = OFF_X  + ACT_BYTES;               // branch temp
static constexpr size_t OFF_CF    = OFF_T  + ACT_BYTES;               // cls features
static constexpr size_t OFF_RF    = OFF_CF + ACT_BYTES;               // reg features
static constexpr size_t OFF_STEMW = OFF_RF + ACT_BYTES;               // 458752 bf16 (3 stems)
static constexpr size_t OFF_CLSW3 = OFF_STEMW + 458752ull * 2;        // 6 packed 3x3 convs
static constexpr size_t OFF_REGW3 = OFF_CLSW3 + 3538944ull * 2;
static constexpr size_t OFF_CLSPW = OFF_REGW3 + 3538944ull * 2;       // 3*80*256 bf16
static constexpr size_t OFF_ROW   = OFF_CLSPW + 61440ull * 2;         // 3*16*256 bf16 (reg+obj, padded)
static constexpr size_t OFF_ROB   = OFF_ROW   + 12288ull * 2;         // 3*16 fp32 bias
static constexpr size_t OFF_ZBUF  = OFF_ROB   + 192;                  // 128B zeros for OOB taps

// ---------------- CDNA5 async global->LDS copy (ASYNCcnt-tracked) ----------------
__device__ __forceinline__ uint32_t lds_off(const void* p) {
    // generic pointer to __shared__: low 32 bits are the wave-relative LDS byte address
    return (uint32_t)(uintptr_t)p;
}
__device__ __forceinline__ void async_copy_b128(uint32_t lds_addr, const void* gaddr) {
    asm volatile("global_load_async_to_lds_b128 %0, %1, off"
                 :: "v"(lds_addr), "v"(gaddr)
                 : "memory");
}
__device__ __forceinline__ void wait_async0() {
    asm volatile("s_wait_asynccnt 0x0" ::: "memory");
}

// ---------------- small prep kernels ----------------
__global__ void zero_bf16_kernel(bf16* p, int n) {
    int i = blockIdx.x * blockDim.x + threadIdx.x;
    if (i < n) p[i] = (bf16)0.0f;
}

__global__ void f32_to_bf16_kernel(const float* __restrict__ src, bf16* __restrict__ dst, int n) {
    int i = blockIdx.x * blockDim.x + threadIdx.x;
    if (i < n) dst[i] = (bf16)src[i];
}

// [B,C,H,W] fp32 -> [B,H,W,C] bf16
__global__ void nchw_to_nhwc_kernel(const float* __restrict__ src, bf16* __restrict__ dst,
                                    int C, int H, int W, int n) {
    int i = blockIdx.x * blockDim.x + threadIdx.x;
    if (i >= n) return;
    int c = i % C; int t = i / C;
    int x = t % W; t /= W;
    int y = t % H; int b = t / H;
    dst[i] = (bf16)src[(((size_t)b * C + c) * H + y) * W + x];
}

// [conv][O=256][I=256][3][3] fp32 -> [conv][O][ (ky*3+kx)*256 + ci ] bf16
__global__ void pack_w3x3_kernel(const float* __restrict__ src, bf16* __restrict__ dst, int n) {
    int i = blockIdx.x * blockDim.x + threadIdx.x;
    if (i >= n) return;
    int conv = i / 589824; int r = i % 589824;
    int o = r / 2304; int k = r % 2304;
    int t = k / 256;  int ci = k % 256;
    int ky = t / 3,   kx = t % 3;
    dst[i] = (bf16)src[(size_t)conv * 589824 + ((size_t)(o * 256 + ci)) * 9 + ky * 3 + kx];
}

// rows 0-3: reg weights, row 4: obj, rows 5-15: zero; plus packed bias
__global__ void pack_regobj_kernel(const float* __restrict__ regw, const float* __restrict__ objw,
                                   const float* __restrict__ regb, const float* __restrict__ objb,
                                   bf16* __restrict__ wout, float* __restrict__ bout) {
    int i = blockIdx.x * blockDim.x + threadIdx.x;      // over 3*16*256
    if (i >= 3 * 16 * 256) return;
    int k = i / (16 * 256); int r = (i / 256) % 16; int c = i % 256;
    float v = 0.0f;
    if (r < 4)       v = regw[((size_t)k * 4 + r) * 256 + c];
    else if (r == 4) v = objw[(size_t)k * 256 + c];
    wout[i] = (bf16)v;
    if (c == 0) {
        float bv = 0.0f;
        if (r < 4)       bv = regb[k * 4 + r];
        else if (r == 4) bv = objb[k];
        bout[k * 16 + r] = bv;
    }
}

// ---------------- WMMA fragment load ----------------
// 16-bit A/B layout: lanes 0-15 take K chunks {0..7, 16..23}, lanes 16-31 take {8..15, 24..31}.
__device__ __forceinline__ v16bf ld_frag(const bf16* p, int hi) {
    const bf16* q = p + (hi ? 8 : 0);
    v8bf lo  = *(const v8bf*)(q);
    v8bf hi8 = *(const v8bf*)(q + 16);
    return __builtin_shufflevector(lo, hi8, 0, 1, 2, 3, 4, 5, 6, 7, 8, 9, 10, 11, 12, 13, 14, 15);
}

__device__ __forceinline__ float silu(float v) { return v / (1.0f + __expf(-v)); }
__device__ __forceinline__ float sigm(float v) { return 1.0f / (1.0f + __expf(-v)); }

// ---------------- LDS-tiled implicit-GEMM conv (+BN +SiLU) ----------------
// Block tile: 128 pixels x 64 couts, 8 waves (4x2), each wave 32x32 (4 WMMA accums).
// Per staging round: 64 K-values of A (128x64, im2col w/ OOB->zbuf) + B (64x64) moved to LDS
// via global_load_async_to_lds_b128 (ASYNCcnt), double buffered -> 8 WMMAs + 16 ds_load_b128
// per round between barriers. Row pitch 144B: 16B-aligned, 16 fragment lanes on distinct banks.
template <int KS>
__global__ __launch_bounds__(256) void conv_gemm_silu(
    const bf16* __restrict__ src, const bf16* __restrict__ wt,
    const float* __restrict__ scale, const float* __restrict__ bias,
    bf16* __restrict__ dst, const bf16* __restrict__ zbuf,
    int H, int W, int Cin, int Cout) {
    __shared__ __align__(16) bf16 sA[2][128][72];
    __shared__ __align__(16) bf16 sB[2][64][72];

    const int tid    = threadIdx.x;
    const int lane   = tid & 31;
    const int wave   = tid >> 5;
    const int hi     = lane >> 4;
    const int m      = lane & 15;
    const int HW     = H * W;
    const int K      = KS * KS * Cin;
    const int NK64   = K >> 6;            // 64-K staging rounds (2304/64=36; stems 4/8/16)

    const int nblkcnt = Cout >> 6;
    const int mblk    = blockIdx.x / nblkcnt;
    const int nblk    = blockIdx.x % nblkcnt;
    const int pbase   = mblk * 128;
    const int nbase   = nblk * 64;

    // staging roles: A has 1024 16B-chunks (4/thread), B has 512 (2/thread)
    const int rowA = tid >> 3;            // 0..31 (rows rowA + 32*j)
    const int chA  = tid & 7;             // 0..7 -> K offset chA*8
    const int rowB = tid >> 3;            // 0..31 (rows rowB + 32*j)
    const int chB  = tid & 7;

    // per-thread pixel coords for the 4 staged A rows (fixed across K loop)
    int ba[4], ya[4], xa[4];
#pragma unroll
    for (int j = 0; j < 4; ++j) {
        int p = pbase + rowA + j * 32;
        ba[j] = p / HW; int rem = p - ba[j] * HW;
        ya[j] = rem / W; xa[j] = rem - ya[j] * W;
    }

    auto stage = [&](int bufn, int ky, int kx, int kb, int ci0) {
#pragma unroll
        for (int j = 0; j < 4; ++j) {
            int r = rowA + j * 32;
            const bf16* g;
            if (KS == 1) {
                g = src + (size_t)(pbase + r) * Cin + ci0 + chA * 8;
            } else {
                int iy = ya[j] + ky - 1, ix = xa[j] + kx - 1;
                bool valid = ((unsigned)iy < (unsigned)H) && ((unsigned)ix < (unsigned)W);
                g = valid ? src + ((size_t)ba[j] * HW + (size_t)iy * W + ix) * Cin + ci0 + chA * 8
                          : zbuf;
            }
            async_copy_b128(lds_off(&sA[bufn][r][chA * 8]), g);
        }
#pragma unroll
        for (int j = 0; j < 2; ++j) {
            int r = rowB + j * 32;
            const bf16* gb = wt + (size_t)(nbase + r) * K + kb + ci0 + chB * 8;
            async_copy_b128(lds_off(&sB[bufn][r][chB * 8]), gb);
        }
    };

    v8f a00 = {}, a01 = {}, a10 = {}, a11 = {};

    // prologue: stage round 0 into buffer 0
    stage(0, 0, 0, 0, 0);
    int ky = 0, kx = 0, kb = 0, ci = 0;   // state advanced before staging the next round

    const int am0 = wave >> 1;            // 0..3 -> A row group
    const int an0 = wave & 1;             // 0..1 -> B row group
    const int arow0 = am0 * 32 + m;
    const int brow0 = an0 * 32 + m;

    for (int ks = 0; ks < NK64; ++ks) {
        wait_async0();
        __syncthreads();                  // staged tiles for round ks visible

        if (ks + 1 < NK64) {
            ci += 64;
            if (ci == Cin) {
                ci = 0; kb += Cin;
                if (KS == 3) { kx++; if (kx == 3) { kx = 0; ky++; } }
            }
            stage((ks + 1) & 1, ky, kx, kb, ci);
        }

        const int buf = ks & 1;
#pragma unroll
        for (int sub = 0; sub < 2; ++sub) {
            const int ko = sub * 32;
            v16bf fa0 = ld_frag(&sA[buf][arow0][ko],      hi);
            v16bf fa1 = ld_frag(&sA[buf][arow0 + 16][ko], hi);
            v16bf fb0 = ld_frag(&sB[buf][brow0][ko],      hi);
            v16bf fb1 = ld_frag(&sB[buf][brow0 + 16][ko], hi);
            a00 = __builtin_amdgcn_wmma_f32_16x16x32_bf16(false, fa0, false, fb0, (short)0, a00, false, false);
            a01 = __builtin_amdgcn_wmma_f32_16x16x32_bf16(false, fa0, false, fb1, (short)0, a01, false, false);
            a10 = __builtin_amdgcn_wmma_f32_16x16x32_bf16(false, fa1, false, fb0, (short)0, a10, false, false);
            a11 = __builtin_amdgcn_wmma_f32_16x16x32_bf16(false, fa1, false, fb1, (short)0, a11, false, false);
        }

        __syncthreads();                  // all reads done before buffer is restaged
    }

    // epilogue: lane holds col n=lane&15, rows (hi ? 8..15 : 0..7) of each 16x16 tile
    const int co_base = nbase + an0 * 32;
    const int pm_base = pbase + am0 * 32;
#pragma unroll
    for (int ni = 0; ni < 2; ++ni) {
        int co = co_base + ni * 16 + m;
        float s  = scale[co];
        float bi = bias[co];
#pragma unroll
        for (int mi = 0; mi < 2; ++mi) {
            v8f acc = (mi == 0) ? (ni == 0 ? a00 : a01) : (ni == 0 ? a10 : a11);
            int pb = pm_base + mi * 16 + hi * 8;
#pragma unroll
            for (int r = 0; r < 8; ++r) {
                float v = acc[r] * s + bi;
                dst[(size_t)(pb + r) * Cout + co] = (bf16)silu(v);
            }
        }
    }
}

// ---------------- prediction GEMM with fused sigmoid/exp/grid decode ----------------
// feat: [Npix][256] bf16; wt: [ntiles*16][256] bf16; out: fp32 [B,8400,85]
__global__ __launch_bounds__(256) void pred_gemm(
    const bf16* __restrict__ feat, const bf16* __restrict__ wt,
    const float* __restrict__ bias, float* __restrict__ out,
    int H, int W, int ntiles, int is_cls, int lvl_off, float stride_f) {
    const int lane = threadIdx.x & 31;
    const int wave = threadIdx.x >> 5;
    const int hi   = lane >> 4;
    const int m    = lane & 15;
    const int HW   = H * W;

    int gw    = blockIdx.x * 8 + wave;
    int mtile = gw / ntiles;
    int nt    = gw % ntiles;

    int p = mtile * 16 + m;
    const bf16* arow = feat + (size_t)p * 256;
    const bf16* wrow = wt + (size_t)(nt * 16 + m) * 256;

    v8f acc = {};
#pragma unroll
    for (int ci = 0; ci < 256; ci += 32) {
        v16bf a  = ld_frag(arow + ci, hi);
        v16bf bb = ld_frag(wrow + ci, hi);
        acc = __builtin_amdgcn_wmma_f32_16x16x32_bf16(false, a, false, bb, (short)0, acc, false, false);
    }

    const int c = nt * 16 + m;
    const float bi = bias[c];
#pragma unroll
    for (int r = 0; r < 8; ++r) {
        int pp = mtile * 16 + hi * 8 + r;
        int b  = pp / HW; int rem = pp - b * HW;
        size_t row = (size_t)b * 8400 + lvl_off + rem;
        float v = acc[r] + bi;
        if (is_cls) {
            out[row * 85 + 5 + c] = sigm(v);                                   // cls sigmoid
        } else {
            int yq = rem / W, xq = rem - yq * W;
            if (c == 0)      out[row * 85 + 0] = (v + (float)xq) * stride_f;   // x decode
            else if (c == 1) out[row * 85 + 1] = (v + (float)yq) * stride_f;   // y decode
            else if (c == 2) out[row * 85 + 2] = __expf(v) * stride_f;         // w decode
            else if (c == 3) out[row * 85 + 3] = __expf(v) * stride_f;         // h decode
            else if (c == 4) out[row * 85 + 4] = sigm(v);                      // obj sigmoid
            // c in 5..15 are zero-padded columns: no store
        }
    }
}

// ---------------- launch ----------------
extern "C" void kernel_launch(void* const* d_in, const int* in_sizes, int n_in,
                              void* d_out, int out_size, void* d_ws, size_t ws_size,
                              hipStream_t stream) {
    (void)in_sizes; (void)n_in; (void)out_size; (void)ws_size;

    const float* x[3]      = {(const float*)d_in[0], (const float*)d_in[1], (const float*)d_in[2]};
    const float* stem_w[3] = {(const float*)d_in[3], (const float*)d_in[4], (const float*)d_in[5]};
    const float* stem_s = (const float*)d_in[6];
    const float* stem_b = (const float*)d_in[7];
    const float* cls_w  = (const float*)d_in[8];
    const float* cls_s  = (const float*)d_in[9];
    const float* cls_b  = (const float*)d_in[10];
    const float* reg_w  = (const float*)d_in[11];
    const float* reg_s  = (const float*)d_in[12];
    const float* reg_b  = (const float*)d_in[13];
    const float* clsp_w = (const float*)d_in[14];
    const float* clsp_b = (const float*)d_in[15];
    const float* regp_w = (const float*)d_in[16];
    const float* regp_b = (const float*)d_in[17];
    const float* objp_w = (const float*)d_in[18];
    const float* objp_b = (const float*)d_in[19];

    char* ws = (char*)d_ws;
    bf16*  IN    = (bf16*)(ws + OFF_IN);
    bf16*  X     = (bf16*)(ws + OFF_X);
    bf16*  T     = (bf16*)(ws + OFF_T);
    bf16*  CF    = (bf16*)(ws + OFF_CF);
    bf16*  RF    = (bf16*)(ws + OFF_RF);
    bf16*  STEMW = (bf16*)(ws + OFF_STEMW);
    bf16*  CLSW3 = (bf16*)(ws + OFF_CLSW3);
    bf16*  REGW3 = (bf16*)(ws + OFF_REGW3);
    bf16*  CLSPW = (bf16*)(ws + OFF_CLSPW);
    bf16*  ROW   = (bf16*)(ws + OFF_ROW);
    float* ROB   = (float*)(ws + OFF_ROB);
    bf16*  ZBUF  = (bf16*)(ws + OFF_ZBUF);

    float* out = (float*)d_out;
    auto cdiv = [](int a, int b) { return (a + b - 1) / b; };

    // ---- weight prep (deterministic, every call) ----
    zero_bf16_kernel<<<1, 64, 0, stream>>>(ZBUF, 64);

    const int stem_elems[3] = {256 * 256, 256 * 512, 256 * 1024};
    const size_t stem_off[3] = {0, 65536, 196608};
    for (int k = 0; k < 3; ++k)
        f32_to_bf16_kernel<<<cdiv(stem_elems[k], 256), 256, 0, stream>>>(stem_w[k], STEMW + stem_off[k], stem_elems[k]);

    f32_to_bf16_kernel<<<cdiv(61440, 256), 256, 0, stream>>>(clsp_w, CLSPW, 61440);
    pack_w3x3_kernel<<<cdiv(3538944, 256), 256, 0, stream>>>(cls_w, CLSW3, 3538944);
    pack_w3x3_kernel<<<cdiv(3538944, 256), 256, 0, stream>>>(reg_w, REGW3, 3538944);
    pack_regobj_kernel<<<cdiv(3 * 16 * 256, 256), 256, 0, stream>>>(regp_w, objp_w, regp_b, objp_b, ROW, ROB);

    // ---- per-level pipeline ----
    const int Hs[3] = {80, 40, 20}, Ws[3] = {80, 40, 20}, Cins[3] = {256, 512, 1024};
    const int lvl_off[3] = {0, 6400, 8000};
    const float strides[3] = {8.0f, 16.0f, 32.0f};

    for (int k = 0; k < 3; ++k) {
        const int H = Hs[k], W = Ws[k], Cin = Cins[k];
        const int Npix = 8 * H * W;
        const int convBlocks = (Npix / 128) * (256 / 64);   // 128px x 64co block tiles

        nchw_to_nhwc_kernel<<<cdiv(Npix * Cin, 256), 256, 0, stream>>>(x[k], IN, Cin, H, W, Npix * Cin);

        // stem 1x1 (Cin -> 256) + SiLU
        conv_gemm_silu<1><<<convBlocks, 256, 0, stream>>>(
            IN, STEMW + stem_off[k], stem_s + k * 256, stem_b + k * 256, X, ZBUF, H, W, Cin, 256);

        // cls branch: two 3x3 convs
        conv_gemm_silu<3><<<convBlocks, 256, 0, stream>>>(
            X, CLSW3 + (size_t)(k * 2 + 0) * 589824, cls_s + (k * 2 + 0) * 256, cls_b + (k * 2 + 0) * 256,
            T, ZBUF, H, W, 256, 256);
        conv_gemm_silu<3><<<convBlocks, 256, 0, stream>>>(
            T, CLSW3 + (size_t)(k * 2 + 1) * 589824, cls_s + (k * 2 + 1) * 256, cls_b + (k * 2 + 1) * 256,
            CF, ZBUF, H, W, 256, 256);

        // reg branch: two 3x3 convs
        conv_gemm_silu<3><<<convBlocks, 256, 0, stream>>>(
            X, REGW3 + (size_t)(k * 2 + 0) * 589824, reg_s + (k * 2 + 0) * 256, reg_b + (k * 2 + 0) * 256,
            T, ZBUF, H, W, 256, 256);
        conv_gemm_silu<3><<<convBlocks, 256, 0, stream>>>(
            T, REGW3 + (size_t)(k * 2 + 1) * 589824, reg_s + (k * 2 + 1) * 256, reg_b + (k * 2 + 1) * 256,
            RF, ZBUF, H, W, 256, 256);

        // predictions + decode, fused into GEMM epilogues
        pred_gemm<<<(Npix / 16) * 5 / 8, 256, 0, stream>>>(
            CF, CLSPW + (size_t)k * 80 * 256, clsp_b + k * 80, out, H, W, 5, 1, lvl_off[k], strides[k]);
        pred_gemm<<<(Npix / 16) / 8, 256, 0, stream>>>(
            RF, ROW + (size_t)k * 16 * 256, ROB + k * 16, out, H, W, 1, 0, lvl_off[k], strides[k]);
    }
}
